// LastFMModel_62912680952376
// MI455X (gfx1250) — compile-verified, matrix operand order checked by reading, not run on hardware
//
#include <hip/hip_runtime.h>
#include <cstddef>

// ---------------- CDNA5 WMMA types ----------------
typedef __attribute__((ext_vector_type(16))) __bf16 v16bf;
typedef __attribute__((ext_vector_type(8)))  float  v8f;

#define ACT_NONE 0
#define ACT_RELU 1
#define ACT_TANH 2

// ---- static problem structure (from reference) ----
#define NU 50000
#define NI 80000
#define NT 10000
#define FEAT 256
#define HID 128
#define INTRA 64
#define INTER 128
#define EDG 500000

// Native f32 -> bf16 conversion: gfx1250 has a hardware BF16 datapath,
// so fptrunc should select the packed cvt instruction instead of the
// 3-VALU-op manual round-to-nearest-even sequence.
__device__ __forceinline__ __bf16 f2bf(float f) { return (__bf16)f; }

// ---------------------------------------------------------------
// GEMM:  C[M,N] = act( rowscale_m * (A[M,K] @ B[K,N]) + bias )
// N, K, LDC compile-time; M runtime.  One wave32 = one 16x16 tile,
// K stepped by 32 via v_wmma_f32_16x16x32_bf16 (f32 accumulate).
// Out-of-range A rows are CLAMPED, not zeroed: in WMMA, A row m only
// feeds D row m and we never store m >= M, so clamped garbage is dead.
// Row-mean (1/deg) is applied in the epilogue: row scaling commutes
// through the matmul, so it leaves the K loop pure load+convert+wmma.
// ---------------------------------------------------------------
template<int N, int K, int LDC, int ACT, bool BIAS, bool ROWDIV>
__global__ __launch_bounds__(256)
void gemm_wmma(const float* __restrict__ A, const float* __restrict__ B,
               const float* __restrict__ bias, const float* __restrict__ rowdiv,
               float* __restrict__ C, int M)
{
    static_assert(K % 32 == 0 && N % 16 == 0, "shape");
    const int lane = threadIdx.x & 31;
    const int wave = threadIdx.x >> 5;
    const int tn   = blockIdx.x;                 // 16-wide N tile
    const int tm   = blockIdx.y * 8 + wave;      // 16-tall M tile (8 waves/block)
    if (tm * 16 >= M) return;                    // wave-uniform exit (EXEC stays full)

    const int half = lane >> 4;                  // 0: lanes 0-15, 1: lanes 16-31
    const int arow = tm * 16 + (lane & 15);
    const int arow_c = (arow < M) ? arow : (M - 1);   // clamp: no conditional loads
    const int ncol = tn * 16 + (lane & 15);           // always < N (N mult of 16)

    const float* Arow = A + (size_t)arow_c * K;

    v8f acc = {0.f, 0.f, 0.f, 0.f, 0.f, 0.f, 0.f, 0.f};

    #pragma unroll
    for (int k0 = 0; k0 < K; k0 += 32) {
        v16bf a, b;
        #pragma unroll
        for (int j = 0; j < 16; ++j) {
            // ISA 7.12.2 16-bit fragment layout: element j -> K = (j/8)*16 + half*8 + (j%8)
            const int kk = k0 + ((j >> 3) << 4) + (half << 3) + (j & 7);
            a[j] = f2bf(Arow[kk]);
            b[j] = f2bf(B[(size_t)kk * N + ncol]);
        }
        if (k0 + 32 < K)
            __builtin_prefetch(Arow + k0 + 32, 0, 3);   // global_prefetch_b8, near
        acc = __builtin_amdgcn_wmma_f32_16x16x32_bf16(
                  false, a, false, b, (short)0, acc, false, false);
    }

    const float bv = BIAS ? bias[ncol] : 0.0f;
    #pragma unroll
    for (int r = 0; r < 8; ++r) {
        const int m = tm * 16 + r + (half << 3);  // C/D layout: VGPR r -> M = r + 8*half
        if (m < M) {
            float v = acc[r];
            if (ROWDIV) {
                float dg = rowdiv[m];
                v *= (dg > 1.0f) ? (1.0f / dg) : 1.0f;   // GraphConv mean (clip(deg,1))
            }
            v += bv;
            if (ACT == ACT_RELU)      v = fmaxf(v, 0.0f);
            else if (ACT == ACT_TANH) v = tanhf(v);
            C[(size_t)m * LDC + ncol] = v;
        }
    }
}

// ---------------- utility kernels ----------------
__global__ void fill_zero(float* __restrict__ p, size_t n) {
    size_t stride = (size_t)gridDim.x * blockDim.x;
    for (size_t i = (size_t)blockIdx.x * blockDim.x + threadIdx.x; i < n; i += stride)
        p[i] = 0.0f;
}

// CompGCN message scatter: agg[dst] += h[src] * comp ; deg[dst] += 1
__global__ void scatter_edges(const float* __restrict__ h, const int* __restrict__ src,
                              const int* __restrict__ dst, const float* __restrict__ comp,
                              float* __restrict__ agg, float* __restrict__ deg, int E)
{
    size_t total  = (size_t)E * HID;
    size_t stride = (size_t)gridDim.x * blockDim.x;
    for (size_t i = (size_t)blockIdx.x * blockDim.x + threadIdx.x; i < total; i += stride) {
        int e = (int)(i >> 7);
        int f = (int)(i & (HID - 1));
        int s = src[e], d = dst[e];
        atomicAdd(&agg[(size_t)d * HID + f], h[(size_t)s * HID + f] * comp[f]);
        if (f == 0) atomicAdd(&deg[d], 1.0f);
    }
}

// out[i,:] = h[perm[i],:]
__global__ void permute_gather(float* __restrict__ out, const float* __restrict__ h,
                               const int* __restrict__ perm, int N)
{
    size_t total  = (size_t)N * HID;
    size_t stride = (size_t)gridDim.x * blockDim.x;
    for (size_t i = (size_t)blockIdx.x * blockDim.x + threadIdx.x; i < total; i += stride) {
        int row = (int)(i >> 7);
        int f   = (int)(i & (HID - 1));
        out[i] = h[(size_t)perm[row] * HID + f];
    }
}

// orth += mean( (z_p^T z_q)^2 ) : one block per (i,j) of the 64x64 Gram matrix
__global__ void orth_pair(const float* __restrict__ z, int N, int p, int q,
                          float* __restrict__ loss)
{
    const int i = blockIdx.x >> 6;
    const int j = blockIdx.x & 63;
    float acc = 0.0f;
    for (int n = threadIdx.x; n < N; n += blockDim.x)
        acc += z[(size_t)n * (3 * INTRA) + p * INTRA + i] *
               z[(size_t)n * (3 * INTRA) + q * INTRA + j];
    __shared__ float red[256];
    red[threadIdx.x] = acc;
    __syncthreads();
    for (int s = 128; s > 0; s >>= 1) {
        if (threadIdx.x < s) red[threadIdx.x] += red[threadIdx.x + s];
        __syncthreads();
    }
    if (threadIdx.x == 0) {
        float g = red[0];
        atomicAdd(loss, g * g * (1.0f / (INTRA * INTRA)));
    }
}

// reconstruction: z_rec = fused @ Vr ; h_rec_p = z_rec[:,p*64:+64] @ fcr[p]
// loss += sum_p mean((h_rec_p - outs_p)^2).  One block per row.
__global__ void recon_loss(const float* __restrict__ fused, const float* __restrict__ Vr,
                           const float* __restrict__ fcr, const float* __restrict__ outs,
                           size_t pstride, int N, float* __restrict__ loss)
{
    const int row = blockIdx.x;
    const int tid = threadIdx.x;
    __shared__ float fr[INTER];
    __shared__ float zr[3 * INTRA];
    __shared__ float red[256];

    if (tid < INTER) fr[tid] = fused[(size_t)row * INTER + tid];
    __syncthreads();
    if (tid < 3 * INTRA) {
        float s = 0.0f;
        #pragma unroll 8
        for (int b = 0; b < INTER; ++b)
            s += fr[b] * Vr[(size_t)b * (3 * INTRA) + tid];
        zr[tid] = s;
    }
    __syncthreads();

    float acc = 0.0f;
    if (tid < HID) {
        #pragma unroll
        for (int p = 0; p < 3; ++p) {
            float s = 0.0f;
            #pragma unroll 8
            for (int a = 0; a < INTRA; ++a)
                s += zr[p * INTRA + a] * fcr[(size_t)(p * INTRA + a) * HID + tid];
            float d = s - outs[(size_t)p * pstride + (size_t)row * HID + tid];
            acc += d * d;
        }
    }
    red[tid] = acc;
    __syncthreads();
    for (int s = 128; s > 0; s >>= 1) {
        if (tid < s) red[tid] += red[tid + s];
        __syncthreads();
    }
    if (tid == 0)
        atomicAdd(loss, red[0] * (1.0f / ((float)N * (float)HID)));
}

// ---------------- host orchestration ----------------
struct HopDef { int st, dt, et; };

extern "C" void kernel_launch(void* const* d_in, const int* in_sizes, int n_in,
                              void* d_out, int out_size, void* d_ws, size_t ws_size,
                              hipStream_t stream)
{
    (void)in_sizes; (void)n_in; (void)out_size; (void)ws_size;

    const float* features   = (const float*)d_in[0];
    const int*   edges_user = (const int*)  d_in[1];
    const int*   edges_item = (const int*)  d_in[2];
    const int*   perm_user  = (const int*)  d_in[3];
    const int*   perm_item  = (const int*)  d_in[4];
    const float* W_type     = (const float*)d_in[5];
    const float* b_type     = (const float*)d_in[6];
    const float* comp_vec   = (const float*)d_in[7];
    const float* W_gcn      = (const float*)d_in[8];
    const float* b_gcn      = (const float*)d_in[9];
    const float* fc_u       = (const float*)d_in[10];
    const float* fcr_u      = (const float*)d_in[11];
    const float* V_u        = (const float*)d_in[12];
    const float* Vr_u       = (const float*)d_in[13];
    const float* fc_i       = (const float*)d_in[14];
    const float* fcr_i      = (const float*)d_in[15];
    const float* V_i        = (const float*)d_in[16];
    const float* Vr_i       = (const float*)d_in[17];

    float* out  = (float*)d_out;
    float* loss = out + (size_t)(NU + NI) * INTER;   // last output element

    static const int CNT[3]  = {NU, NI, NT};
    static const int OFFS[3] = {0, NU, NU + NI};
    static const int GRP[5]  = {2, 1, 1, 2, 1};

    static const HopDef HOPS_U[7] = {{0,1,0},{1,0,1},{0,1,0},{1,2,2},{2,1,3},{1,0,1},{0,0,4}};
    static const int    PATH_U[4] = {0, 2, 6, 7};
    static const HopDef HOPS_I[7] = {{1,0,1},{0,1,0},{1,2,2},{2,1,3},{1,0,1},{0,0,4},{0,1,0}};
    static const int    PATH_I[4] = {0, 2, 4, 7};

    // ---- workspace carve (floats) ----
    float* ws = (float*)d_ws;
    size_t o = 0;
    float* tfeats = ws + o; o += (size_t)(NU + NI + NT) * HID;   // 140000 x 128
    float* bufAgg = ws + o; o += (size_t)NI * HID;               // 80000 x 128
    float* bufA   = ws + o; o += (size_t)NI * HID;
    float* bufB   = ws + o; o += (size_t)NI * HID;
    float* deg    = ws + o; o += (size_t)NI;
    const size_t OUT_STRIDE = (size_t)NI * HID;
    float* outsBuf = ws + o; o += 3 * OUT_STRIDE;                // 3 views
    float* zBuf    = ws + o; o += (size_t)NI * (3 * INTRA);      // [N,192]

    auto grid_for = [](int M, int N) { return dim3((unsigned)(N / 16), (unsigned)((M + 127) / 128)); };

    fill_zero<<<1, 64, 0, stream>>>(loss, 1);

    // ---- NodeTransform: t_feats[t] = relu(feat_t @ W_type[t] + b_type[t]) ----
    for (int t = 0; t < 3; ++t)
        gemm_wmma<HID, FEAT, HID, ACT_RELU, true, false>
            <<<grid_for(CNT[t], HID), 256, 0, stream>>>(
                features + (size_t)OFFS[t] * FEAT,
                W_type + (size_t)t * FEAT * HID, b_type + (size_t)t * HID,
                nullptr, tfeats + (size_t)OFFS[t] * HID, CNT[t]);

    // ---- two metapath layers ----
    for (int layer = 0; layer < 2; ++layer) {
        const HopDef* hops = layer ? HOPS_I : HOPS_U;
        const int*    path = layer ? PATH_I : PATH_U;
        const int*    edges = layer ? edges_item : edges_user;
        const int*    perm  = layer ? perm_item  : perm_user;
        const int     Nout  = layer ? NI : NU;
        const float*  fc  = layer ? fc_i  : fc_u;
        const float*  fcr = layer ? fcr_i : fcr_u;
        const float*  Vm  = layer ? V_i   : V_u;
        const float*  Vr  = layer ? Vr_i  : Vr_u;
        float* outp = out + (layer ? (size_t)NU * INTER : 0);

        for (int pi = 0; pi < 3; ++pi) {
            const float* hcur = tfeats + (size_t)OFFS[hops[path[pi]].st] * HID;
            int flip = 0;
            for (int k = path[pi]; k < path[pi + 1]; ++k) {
                const HopDef hd = hops[k];
                const int nd = CNT[hd.dt];
                fill_zero<<<2048, 256, 0, stream>>>(bufAgg, (size_t)nd * HID);
                fill_zero<<<64,   256, 0, stream>>>(deg, (size_t)nd);
                const int* srcp = edges + (size_t)k * 2 * EDG;
                const int* dstp = srcp + EDG;
                scatter_edges<<<4096, 256, 0, stream>>>(
                    hcur, srcp, dstp, comp_vec + (size_t)hd.et * HID, bufAgg, deg, EDG);
                const int g = GRP[hd.et];
                float* hnext = flip ? bufB : bufA;
                // h = (agg @ W_gcn[g]) / clip(deg,1) + b_gcn[g]   (mean in epilogue)
                gemm_wmma<HID, HID, HID, ACT_NONE, true, true>
                    <<<grid_for(nd, HID), 256, 0, stream>>>(
                        bufAgg, W_gcn + (size_t)g * HID * HID,
                        b_gcn + (size_t)g * HID, deg, hnext, nd);
                hcur = hnext;
                flip ^= 1;
            }
            permute_gather<<<2048, 256, 0, stream>>>(
                outsBuf + (size_t)pi * OUT_STRIDE, hcur, perm, Nout);
        }

        // ---- fusion: z_p = tanh(outs_p @ fc[p]) packed into [N,192] ----
        for (int p = 0; p < 3; ++p)
            gemm_wmma<INTRA, HID, 3 * INTRA, ACT_TANH, false, false>
                <<<grid_for(Nout, INTRA), 256, 0, stream>>>(
                    outsBuf + (size_t)p * OUT_STRIDE,
                    fc + (size_t)p * HID * INTRA, nullptr, nullptr,
                    zBuf + (size_t)p * INTRA, Nout);

        // fused = z @ V  -> directly into d_out segment
        gemm_wmma<INTER, 3 * INTRA, INTER, ACT_NONE, false, false>
            <<<grid_for(Nout, INTER), 256, 0, stream>>>(
                zBuf, Vm, nullptr, nullptr, outp, Nout);

        // orthogonality + reconstruction losses
        for (int p = 0; p < 3; ++p)
            for (int q = p + 1; q < 3; ++q)
                orth_pair<<<INTRA * INTRA, 256, 0, stream>>>(zBuf, Nout, p, q, loss);
        recon_loss<<<Nout, 256, 0, stream>>>(outp, Vr, fcr, outsBuf,
                                             OUT_STRIDE, Nout, loss);
    }
}
